// NemotronHMamba2Mixer_4174708212186
// MI455X (gfx1250) — matrix operand order repaired and checked by reading, not compiled, not run on hardware
//
#include <hip/hip_runtime.h>
#include <hip/hip_bf16.h>
#include <cstdint>

typedef __attribute__((ext_vector_type(16))) __bf16       v16bf;
typedef __attribute__((ext_vector_type(8)))  float        v8f;
typedef __attribute__((ext_vector_type(4)))  unsigned int v4u;

#define DEV __device__ __forceinline__

// ---------------- problem constants ----------------
constexpr int HID   = 4096;
constexpr int NH    = 128;
constexpr int HD    = 64;
constexpr int DIN   = NH * HD;            // 8192
constexpr int G     = 8;
constexpr int NST   = 128;                // state size N
constexpr int CS    = 256;                // chunk size
constexpr int SEQ   = 2048;
constexpr int BATCH = 2;
constexpr int TOK   = BATCH * SEQ;        // 4096
constexpr int CONVD = DIN + 2 * G * NST;  // 10240
constexpr int PROJ  = DIN + CONVD + NH;   // 18560
constexpr int NC    = SEQ / CS;           // 8
constexpr int GRP   = DIN / G;            // 1024
constexpr float EPSV = 1e-5f;

// ---------------- bf16 helpers ----------------
DEV float bf2f(__bf16 h) {
  unsigned short u = __builtin_bit_cast(unsigned short, h);
  unsigned int v = ((unsigned int)u) << 16;
  return __builtin_bit_cast(float, v);
}
DEV __bf16 f2bf(float f) {
  unsigned int u = __builtin_bit_cast(unsigned int, f);
  u += 0x7fffu + ((u >> 16) & 1u);           // round-to-nearest-even
  unsigned short s = (unsigned short)(u >> 16);
  return __builtin_bit_cast(__bf16, s);
}

// ---------------- WMMA fragment loads (ISA 7.12.2 layouts) ----------------
DEV v16bf load_afrag(const __bf16* p, int stride) {
  int lane = threadIdx.x & 31;
  const __bf16* q = p + (size_t)(lane & 15) * stride + ((lane < 16) ? 0 : 8);
  union { v4u u[2]; v16bf v; } x;
  x.u[0] = *(const v4u*)q;
  x.u[1] = *(const v4u*)(q + 16);
  return x.v;
}
DEV v16bf load_bfrag(const __bf16* p, int stride) {
  int lane = threadIdx.x & 31;
  const __bf16* q = p + (size_t)(lane & 15) * stride + ((lane < 16) ? 0 : 8);
  union { v4u u[2]; v16bf v; } x;
  x.u[0] = *(const v4u*)q;
  x.u[1] = *(const v4u*)(q + 16);
  return x.v;
}
DEV v8f wmma_bf(v16bf a, v16bf b, v8f c) {
  return __builtin_amdgcn_wmma_f32_16x16x32_bf16(false, a, false, b, (short)0, c,
                                                 false, false);
}

// ---------------- CDNA5 async global->LDS (ASYNCcnt path) ----------------
DEV void async_ld_b128(unsigned ldsoff, const void* gptr) {
  asm volatile("global_load_async_to_lds_b128 %0, %1, off"
               :: "v"(ldsoff), "v"(gptr)
               : "memory");
}
DEV void wait_async0() {
  asm volatile("s_wait_asynccnt 0x0" ::: "memory");
}

// ---------------- elementwise cast fp32 -> bf16 ----------------
__global__ void k_cast(const float* __restrict__ src, __bf16* __restrict__ dst,
                       size_t n) {
  size_t i = (size_t)blockIdx.x * 256 + threadIdx.x;
  if (i < n) dst[i] = f2bf(src[i]);
}

// ---------------- bf16 WMMA GEMM: C[M,N] = A[M,K] * B[N,K]^T ----------------
// grid: (N/128, M/128), block: 256 threads = 8 waves; double-buffered LDS filled
// by global_load_async_to_lds_b128, one barrier per k-step, copy/compute overlap
__global__ __launch_bounds__(256) void k_gemm(const __bf16* __restrict__ A,
                                              const __bf16* __restrict__ B,
                                              float* __restrict__ C, int K,
                                              int Nld) {
  __shared__ __bf16 As[2][128 * 32];
  __shared__ __bf16 Bs[2][128 * 32];
  const int n0 = blockIdx.x * 128, m0 = blockIdx.y * 128;
  const int t = threadIdx.x, wave = t >> 5, lane = t & 31;

  // per-thread slice of the 8KB tile: elements e0 (+2048 for second half)
  const int e0 = t * 8;
  const int row0 = e0 >> 5, kc0 = e0 & 31;
  const unsigned abase = (unsigned)(size_t)(void*)&As[0][0] + (unsigned)e0 * 2;
  const unsigned bbase = (unsigned)(size_t)(void*)&Bs[0][0] + (unsigned)e0 * 2;

  v8f acc[8] = {};
  const int nIter = K >> 5;

  // prologue: async-copy tile 0 into buffer 0
  {
    const __bf16* ga = A + (size_t)(m0 + row0) * K + kc0;
    const __bf16* gb = B + (size_t)(n0 + row0) * K + kc0;
    async_ld_b128(abase, ga);
    async_ld_b128(abase + 4096u, ga + (size_t)64 * K);
    async_ld_b128(bbase, gb);
    async_ld_b128(bbase + 4096u, gb + (size_t)64 * K);
  }

  for (int it = 0; it < nIter; ++it) {
    wait_async0();        // current buffer's DMA done (this wave)
    __syncthreads();      // all waves' DMA visible; prev compute finished
    if (it + 1 < nIter) { // overlap: DMA next tile while computing current
      const int k0 = (it + 1) << 5;
      const unsigned bo = (unsigned)((it + 1) & 1) * 8192u;
      const __bf16* ga = A + (size_t)(m0 + row0) * K + k0 + kc0;
      const __bf16* gb = B + (size_t)(n0 + row0) * K + k0 + kc0;
      async_ld_b128(abase + bo, ga);
      async_ld_b128(abase + bo + 4096u, ga + (size_t)64 * K);
      async_ld_b128(bbase + bo, gb);
      async_ld_b128(bbase + bo + 4096u, gb + (size_t)64 * K);
    }
    const int buf = it & 1;
    v16bf af = load_afrag(&As[buf][(wave * 16) * 32], 32);
    v16bf bfr[8];
#pragma unroll
    for (int nt = 0; nt < 8; ++nt)
      bfr[nt] = load_bfrag(&Bs[buf][(nt * 16) * 32], 32);
#pragma unroll
    for (int nt = 0; nt < 8; ++nt) acc[nt] = wmma_bf(af, bfr[nt], acc[nt]);
  }
  const int r0 = (lane >> 4) * 8, cn = lane & 15;
#pragma unroll
  for (int nt = 0; nt < 8; ++nt)
#pragma unroll
    for (int r = 0; r < 8; ++r)
      C[(size_t)(m0 + wave * 16 + r0 + r) * Nld + n0 + nt * 16 + cn] = acc[nt][r];
}

// ---------------- causal depthwise conv (K=4) + bias + SiLU ----------------
__global__ void k_conv(const float* __restrict__ zx, const float* __restrict__ cw,
                       const float* __restrict__ cb, float* __restrict__ xbc) {
  size_t idx = (size_t)blockIdx.x * 256 + threadIdx.x;
  if (idx >= (size_t)TOK * CONVD) return;
  int ch = (int)(idx % CONVD);
  size_t tokid = idx / CONVD;
  int l = (int)(tokid % SEQ);
  int b = (int)(tokid / SEQ);
  float acc = cb[ch];
#pragma unroll
  for (int j = 0; j < 4; ++j) {
    int ll = l - 3 + j;
    if (ll >= 0)
      acc += cw[ch * 4 + j] * zx[((size_t)b * SEQ + ll) * PROJ + DIN + ch];
  }
  xbc[idx] = acc / (1.0f + __expf(-acc));
}

// ---------------- dt softplus + per-chunk cumsum of A*dt ----------------
__global__ __launch_bounds__(256) void k_dt(const float* __restrict__ zx,
                                            const float* __restrict__ dtb,
                                            const float* __restrict__ alog,
                                            float* __restrict__ dtsp,
                                            float* __restrict__ cumG) {
  __shared__ float sc[CS];
  const int h = blockIdx.x & (NH - 1);
  const int c = (blockIdx.x >> 7) & (NC - 1);
  const int b = blockIdx.x >> 10;
  const int t = threadIdx.x;
  size_t tok = (size_t)b * SEQ + c * CS + t;
  float raw = zx[tok * PROJ + DIN + CONVD + h] + dtb[h];
  float sp = (raw > 20.0f) ? raw : log1pf(__expf(raw));
  dtsp[tok * NH + h] = sp;
  float v = -__expf(alog[h]) * sp;
  sc[t] = v;
  __syncthreads();
  for (int off = 1; off < CS; off <<= 1) {
    float u = (t >= off) ? sc[t - off] : 0.0f;
    __syncthreads();
    v += u;
    sc[t] = v;
    __syncthreads();
  }
  cumG[((size_t)(b * NC + c) * NH + h) * CS + t] = v;
}

// ---------------- prep: xs^T (d-major), decayed xs^T ----------------
__global__ void k_prep_xs(const float* __restrict__ xbc,
                          const float* __restrict__ dtsp,
                          const float* __restrict__ cumG, __bf16* __restrict__ xs,
                          __bf16* __restrict__ xsd) {
  size_t idx = (size_t)blockIdx.x * 256 + threadIdx.x;
  if (idx >= (size_t)BATCH * NC * NH * HD * CS) return;
  int k = (int)(idx & 255);
  int d = (int)((idx >> 8) & 63);
  int h = (int)((idx >> 14) & 127);
  int c = (int)((idx >> 21) & 7);
  int b = (int)(idx >> 24);
  size_t tok = (size_t)b * SEQ + c * CS + k;
  float v = xbc[tok * CONVD + h * HD + d] * dtsp[tok * NH + h];
  size_t cb = ((size_t)(b * NC + c) * NH + h) * CS;
  float dec = __expf(cumG[cb + (CS - 1)] - cumG[cb + k]);
  xs[idx] = f2bf(v);
  xsd[idx] = f2bf(v * dec);
}

// ---------------- prep: B/C natural bf16 + B^T (n-major) ----------------
__global__ void k_prep_bc(const float* __restrict__ xbc, __bf16* __restrict__ Bn,
                          __bf16* __restrict__ Cn, __bf16* __restrict__ Bt) {
  size_t idx = (size_t)blockIdx.x * 256 + threadIdx.x;
  if (idx >= (size_t)TOK * 2 * G * NST) return;
  int n = (int)(idx & 127);
  int g = (int)((idx >> 7) & 7);
  int s = (int)((idx >> 10) & 1);
  size_t tokid = idx >> 11;
  int l = (int)(tokid & (SEQ - 1));
  int b = (int)(tokid >> 11);
  __bf16 bv = f2bf(xbc[tokid * CONVD + DIN + s * (G * NST) + g * NST + n]);
  size_t o = (tokid * G + g) * NST + n;
  if (s == 0) {
    Bn[o] = bv;
    int c = l >> 8, k = l & (CS - 1);
    Bt[(((size_t)(b * NC + c) * G + g) * NST + n) * CS + k] = bv;
  } else {
    Cn[o] = bv;
  }
}

// ---------------- intra-chunk SSM: G*L in LDS, Y_diag, local states ----------
__global__ __launch_bounds__(256) void k_ssm_intra(
    const __bf16* __restrict__ Bn, const __bf16* __restrict__ Cn,
    const __bf16* __restrict__ xs, const __bf16* __restrict__ xsd,
    const __bf16* __restrict__ Bt, const float* __restrict__ cumG,
    float* __restrict__ ysum, float* __restrict__ sloc) {
  constexpr int GLS = 264;  // padded row stride (16B-aligned)
  __shared__ __bf16 GL[CS * GLS];
  __shared__ float cumS[CS];
  const int h = blockIdx.x & (NH - 1);
  const int c = (blockIdx.x >> 7) & (NC - 1);
  const int b = blockIdx.x >> 10;
  const int g = h >> 4;
  const int t = threadIdx.x, wave = t >> 5, lane = t & 31;

  cumS[t] = cumG[((size_t)(b * NC + c) * NH + h) * CS + t];
  for (int i = t; i < CS * GLS; i += 256) GL[i] = f2bf(0.0f);
  __syncthreads();

  const size_t tokBase = (size_t)b * SEQ + c * CS;
  const int r0 = (lane >> 4) * 8, cl = lane & 15;

  // ---- phase 1: GL[i][j] = (C B^T)[i][j] * exp(cum[i]-cum[j]) (lower tri) ----
  for (int tid = wave; tid < 136; tid += 8) {
    int ti = 0, s = 0;
    while (s + ti + 1 <= tid) { s += ti + 1; ++ti; }
    int tj = tid - s;
    const __bf16* Ca = Cn + ((tokBase + ti * 16) * G + g) * NST;
    const __bf16* Ba = Bn + ((tokBase + tj * 16) * G + g) * NST;
    v8f acc = {};
#pragma unroll
    for (int ks = 0; ks < 4; ++ks) {
      v16bf af = load_afrag(Ca + ks * 32, G * NST);
      v16bf bf = load_bfrag(Ba + ks * 32, G * NST);
      acc = wmma_bf(af, bf, acc);
    }
    int j = tj * 16 + cl;
    float cj = cumS[j];
#pragma unroll
    for (int r = 0; r < 8; ++r) {
      int i = ti * 16 + r0 + r;
      float v = (j <= i) ? acc[r] * __expf(cumS[i] - cj) : 0.0f;
      GL[i * GLS + j] = f2bf(v);
    }
  }
  __syncthreads();

  // ---- phase 2: Y_diag = GL @ xs ----
  const __bf16* xsb = xs + (((size_t)(b * NC + c) * NH + h) * HD) * CS;
  for (int ii = 0; ii < 2; ++ii) {
    const int it = wave * 2 + ii;
    v8f acc[4] = {};
    for (int js = 0; js < 8; ++js) {
      v16bf af = load_afrag(&GL[(it * 16) * GLS + js * 32], GLS);
      v16bf bfr[4];
#pragma unroll
      for (int dt0 = 0; dt0 < 4; ++dt0)
        bfr[dt0] = load_bfrag(xsb + (size_t)(dt0 * 16) * CS + js * 32, CS);
#pragma unroll
      for (int dt0 = 0; dt0 < 4; ++dt0)
        acc[dt0] = wmma_bf(af, bfr[dt0], acc[dt0]);
    }
#pragma unroll
    for (int dt0 = 0; dt0 < 4; ++dt0)
#pragma unroll
      for (int r = 0; r < 8; ++r) {
        int i = it * 16 + r0 + r;
        ysum[(tokBase + i) * DIN + h * HD + dt0 * 16 + cl] = acc[dt0][r];
      }
  }

  // ---- phase 3: S[n,d] = B^T @ (decay*xs) ----
  const __bf16* Bta = Bt + (((size_t)(b * NC + c) * G + g) * NST + wave * 16) * CS;
  const __bf16* xsdb = xsd + (((size_t)(b * NC + c) * NH + h) * HD) * CS;
  v8f sacc[4] = {};
  for (int ks = 0; ks < 8; ++ks) {
    v16bf af = load_afrag(Bta + ks * 32, CS);
    v16bf bfr[4];
#pragma unroll
    for (int dt0 = 0; dt0 < 4; ++dt0)
      bfr[dt0] = load_bfrag(xsdb + (size_t)(dt0 * 16) * CS + ks * 32, CS);
#pragma unroll
    for (int dt0 = 0; dt0 < 4; ++dt0)
      sacc[dt0] = wmma_bf(af, bfr[dt0], sacc[dt0]);
  }
  size_t sbase = ((size_t)(b * NC + c) * NH + h) * NST;
#pragma unroll
  for (int dt0 = 0; dt0 < 4; ++dt0)
#pragma unroll
    for (int r = 0; r < 8; ++r)
      sloc[(sbase + wave * 16 + r0 + r) * HD + dt0 * 16 + cl] = sacc[dt0][r];
}

// ---------------- inter-chunk recurrence: prefix states (transposed to [d][n])
__global__ void k_recur(const float* __restrict__ sloc,
                        const float* __restrict__ cumG, __bf16* __restrict__ Pb) {
  size_t idx = (size_t)blockIdx.x * 256 + threadIdx.x;
  if (idx >= (size_t)BATCH * NH * HD * NST) return;
  int n = (int)(idx & 127);
  int d = (int)((idx >> 7) & 63);
  int h = (int)((idx >> 13) & 127);
  int b = (int)(idx >> 20);
  float P = 0.0f;
  for (int c = 0; c < NC; ++c) {
    size_t base = (size_t)(b * NC + c) * NH + h;
    Pb[(base * HD + d) * NST + n] = f2bf(P);
    float tail = cumG[base * CS + (CS - 1)];
    float S = sloc[(base * NST + n) * HD + d];
    P = __expf(tail) * P + S;
  }
}

// ---------------- Y_off = diag(e^cum) C P^T, + D residual + SiLU gate ---------
__global__ __launch_bounds__(256) void k_ssm_off(const __bf16* __restrict__ Cn,
                                                 const __bf16* __restrict__ Pb,
                                                 const float* __restrict__ cumG,
                                                 const float* __restrict__ xbc,
                                                 const float* __restrict__ zx,
                                                 const float* __restrict__ Dp,
                                                 float* __restrict__ ysum) {
  __shared__ float cumS[CS];
  const int h = blockIdx.x & (NH - 1);
  const int c = (blockIdx.x >> 7) & (NC - 1);
  const int b = blockIdx.x >> 10;
  const int g = h >> 4;
  const int t = threadIdx.x, wave = t >> 5, lane = t & 31;
  cumS[t] = cumG[((size_t)(b * NC + c) * NH + h) * CS + t];
  __syncthreads();

  const size_t tokBase = (size_t)b * SEQ + c * CS;
  const __bf16* Pbase = Pb + (((size_t)(b * NC + c) * NH + h) * HD) * NST;
  const float Dh = Dp[h];
  const int r0 = (lane >> 4) * 8, cl = lane & 15;

  for (int ii = 0; ii < 2; ++ii) {
    const int it = wave * 2 + ii;
    const __bf16* Ca = Cn + ((tokBase + it * 16) * G + g) * NST;
    v8f acc[4] = {};
#pragma unroll
    for (int ks = 0; ks < 4; ++ks) {
      v16bf af = load_afrag(Ca + ks * 32, G * NST);
      v16bf bfr[4];
#pragma unroll
      for (int dt0 = 0; dt0 < 4; ++dt0)
        bfr[dt0] = load_bfrag(Pbase + (size_t)(dt0 * 16) * NST + ks * 32, NST);
#pragma unroll
      for (int dt0 = 0; dt0 < 4; ++dt0)
        acc[dt0] = wmma_bf(af, bfr[dt0], acc[dt0]);
    }
#pragma unroll
    for (int dt0 = 0; dt0 < 4; ++dt0)
#pragma unroll
      for (int r = 0; r < 8; ++r) {
        int i = it * 16 + r0 + r;
        int ch = h * HD + dt0 * 16 + cl;
        size_t tok = tokBase + i;
        float sdec = __expf(cumS[i]);
        size_t yi = tok * DIN + ch;
        float yv = ysum[yi] + acc[dt0][r] * sdec + Dh * xbc[tok * CONVD + ch];
        float gt = zx[tok * PROJ + ch];
        ysum[yi] = yv * gt / (1.0f + __expf(-gt));
      }
  }
}

// ---------------- grouped RMSNorm + norm_w, emit bf16 for out_proj ------------
__global__ __launch_bounds__(256) void k_norm(const float* __restrict__ y,
                                              const float* __restrict__ nw,
                                              __bf16* __restrict__ outb) {
  __shared__ float red[256];
  const size_t tok = blockIdx.x;
  const int t = threadIdx.x;
  for (int g = 0; g < G; ++g) {
    int ch = g * GRP + t * 4;
    float4 v = *(const float4*)&y[tok * DIN + ch];
    red[t] = v.x * v.x + v.y * v.y + v.z * v.z + v.w * v.w;
    __syncthreads();
    for (int s = 128; s > 0; s >>= 1) {
      if (t < s) red[t] += red[t + s];
      __syncthreads();
    }
    float rs = rsqrtf(red[0] / (float)GRP + EPSV);
    __syncthreads();
    outb[tok * DIN + ch + 0] = f2bf(v.x * rs * nw[ch + 0]);
    outb[tok * DIN + ch + 1] = f2bf(v.y * rs * nw[ch + 1]);
    outb[tok * DIN + ch + 2] = f2bf(v.z * rs * nw[ch + 2]);
    outb[tok * DIN + ch + 3] = f2bf(v.w * rs * nw[ch + 3]);
  }
}

// ---------------- workspace layout (bytes) ----------------
constexpr size_t OFF_W1  = 0;                                       // in_proj bf16
constexpr size_t OFF_W2  = OFF_W1 + (size_t)PROJ * HID * 2;         // out_proj bf16
constexpr size_t OFF_HSB = OFF_W2 + (size_t)HID * DIN * 2;          // hs bf16
constexpr size_t OFF_ZX  = OFF_HSB + (size_t)TOK * HID * 2;         // zxbcdt f32
constexpr size_t OFF_XBC = OFF_ZX + (size_t)TOK * PROJ * 4;         // conv out f32
constexpr size_t OFF_DT  = OFF_XBC + (size_t)TOK * CONVD * 4;       // dt softplus
constexpr size_t OFF_CUM = OFF_DT + (size_t)TOK * NH * 4;           // A cumsum
constexpr size_t OFF_XS  = OFF_CUM + (size_t)BATCH * NC * NH * CS * 4;
constexpr size_t OFF_XSD = OFF_XS + (size_t)BATCH * NC * NH * HD * CS * 2;
constexpr size_t OFF_BT  = OFF_XSD + (size_t)BATCH * NC * NH * HD * CS * 2;
constexpr size_t OFF_BN  = OFF_BT + (size_t)BATCH * NC * G * NST * CS * 2;
constexpr size_t OFF_CN  = OFF_BN + (size_t)TOK * G * NST * 2;
constexpr size_t OFF_YS  = OFF_CN + (size_t)TOK * G * NST * 2;      // y accum f32
constexpr size_t OFF_SL  = OFF_YS + (size_t)TOK * DIN * 4;          // local states
constexpr size_t OFF_PB  = OFF_SL + (size_t)BATCH * NC * NH * NST * HD * 4;
constexpr size_t OFF_YN  = OFF_PB + (size_t)BATCH * NC * NH * HD * NST * 2;

static inline unsigned blk(size_t n) { return (unsigned)((n + 255) / 256); }

extern "C" void kernel_launch(void* const* d_in, const int* in_sizes, int n_in,
                              void* d_out, int out_size, void* d_ws,
                              size_t ws_size, hipStream_t stream) {
  (void)in_sizes; (void)n_in; (void)out_size; (void)ws_size;
  const float* hs   = (const float*)d_in[0];
  const float* w1   = (const float*)d_in[1];
  const float* cw   = (const float*)d_in[2];
  const float* cb   = (const float*)d_in[3];
  const float* dtb  = (const float*)d_in[4];
  const float* alog = (const float*)d_in[5];
  const float* Dp   = (const float*)d_in[6];
  const float* nw   = (const float*)d_in[7];
  const float* w2   = (const float*)d_in[8];
  float* out = (float*)d_out;

  char* ws = (char*)d_ws;
  __bf16* w1b  = (__bf16*)(ws + OFF_W1);
  __bf16* w2b  = (__bf16*)(ws + OFF_W2);
  __bf16* hsb  = (__bf16*)(ws + OFF_HSB);
  float*  zx   = (float*)(ws + OFF_ZX);
  float*  xbc  = (float*)(ws + OFF_XBC);
  float*  dtsp = (float*)(ws + OFF_DT);
  float*  cum  = (float*)(ws + OFF_CUM);
  __bf16* xs   = (__bf16*)(ws + OFF_XS);
  __bf16* xsd  = (__bf16*)(ws + OFF_XSD);
  __bf16* Bt   = (__bf16*)(ws + OFF_BT);
  __bf16* Bn   = (__bf16*)(ws + OFF_BN);
  __bf16* Cn   = (__bf16*)(ws + OFF_CN);
  float*  ys   = (float*)(ws + OFF_YS);
  float*  sl   = (float*)(ws + OFF_SL);
  __bf16* Pb   = (__bf16*)(ws + OFF_PB);
  __bf16* ynb  = (__bf16*)(ws + OFF_YN);

  // 1. casts
  k_cast<<<blk((size_t)PROJ * HID), 256, 0, stream>>>(w1, w1b, (size_t)PROJ * HID);
  k_cast<<<blk((size_t)HID * DIN), 256, 0, stream>>>(w2, w2b, (size_t)HID * DIN);
  k_cast<<<blk((size_t)TOK * HID), 256, 0, stream>>>(hs, hsb, (size_t)TOK * HID);

  // 2. in_proj GEMM: zxbcdt = hs @ W1^T
  k_gemm<<<dim3(PROJ / 128, TOK / 128), 256, 0, stream>>>(hsb, w1b, zx, HID, PROJ);

  // 3. conv + silu; dt softplus + A*dt chunk cumsum
  k_conv<<<blk((size_t)TOK * CONVD), 256, 0, stream>>>(zx, cw, cb, xbc);
  k_dt<<<BATCH * NC * NH, 256, 0, stream>>>(zx, dtb, alog, dtsp, cum);

  // 4. operand prep
  k_prep_xs<<<blk((size_t)BATCH * NC * NH * HD * CS), 256, 0, stream>>>(
      xbc, dtsp, cum, xs, xsd);
  k_prep_bc<<<blk((size_t)TOK * 2 * G * NST), 256, 0, stream>>>(xbc, Bn, Cn, Bt);

  // 5. intra-chunk: Y_diag + local states
  k_ssm_intra<<<BATCH * NC * NH, 256, 0, stream>>>(Bn, Cn, xs, xsd, Bt, cum, ys,
                                                   sl);

  // 6. inter-chunk recurrence
  k_recur<<<blk((size_t)BATCH * NH * HD * NST), 256, 0, stream>>>(sl, cum, Pb);

  // 7. Y_off + D residual + gate
  k_ssm_off<<<BATCH * NC * NH, 256, 0, stream>>>(Cn, Pb, cum, xbc, zx, Dp, ys);

  // 8. grouped RMSNorm -> bf16
  k_norm<<<TOK, 256, 0, stream>>>(ys, nw, ynb);

  // 9. out_proj GEMM -> fp32 output
  k_gemm<<<dim3(HID / 128, TOK / 128), 256, 0, stream>>>(ynb, w2b, out, DIN, HID);
}
